// ReacFormerSE3Block_38732015075528
// MI455X (gfx1250) — compile-verified, matrix operand order checked by reading, not run on hardware
//
#include <hip/hip_runtime.h>
#include <hip/hip_bf16.h>

// ---------------------------------------------------------------------------
// Types for CDNA5 WMMA (wave32, 16x16x32 bf16 -> f32)
// ---------------------------------------------------------------------------
typedef __attribute__((ext_vector_type(16))) __bf16 v16bf;
typedef __attribute__((ext_vector_type(8)))  float  v8f;

union BF16x16 { v16bf v; unsigned short u[16]; uint4 q[2]; };

__device__ inline unsigned short f2bf(float x) {
  __bf16 h = (__bf16)x;                       // RNE fptrunc; hw cvt if available
  return __builtin_bit_cast(unsigned short, h);
}

__device__ inline float fast_rcp(float x) { return __builtin_amdgcn_rcpf(x); }

__device__ inline float fast_tanh(float x) {
#if __has_builtin(__builtin_amdgcn_tanhf)
  return __builtin_amdgcn_tanhf(x);
#else
  return tanhf(x);
#endif
}

__device__ inline void wait_async0() {
#if __has_builtin(__builtin_amdgcn_s_wait_asynccnt)
  __builtin_amdgcn_s_wait_asynccnt(0);
#else
  asm volatile("s_wait_asynccnt 0" ::: "memory");
#endif
}

// Async 16B global->LDS copy (gfx1250, tracked by ASYNCcnt).
__device__ inline void async_g2l_b128(void* lds_dst, const void* gsrc) {
  unsigned loff = (unsigned)(size_t)lds_dst;           // low 32b = LDS offset
  unsigned long long ga = (unsigned long long)gsrc;
  asm volatile("global_load_async_to_lds_b128 %0, %1, off"
               :: "v"(loff), "v"(ga) : "memory");
}

__device__ inline v8f wmma_bf16(v16bf a, v16bf b, v8f c) {
  return __builtin_amdgcn_wmma_f32_16x16x32_bf16(false, a, false, b, (short)0, c,
                                                 false, false);
}

// A fragment: 16x32 tile, f32 source row-major (lda elems), convert to bf16.
// Lane L: row = L&15; lanes<16 hold K={0..7,16..23}, lanes>=16 K={8..15,24..31}.
__device__ inline v16bf load_a_f32(const float* __restrict__ A0, int lda, int k0,
                                   int lane) {
  int row = lane & 15, koff = (lane >> 4) * 8;
  const float* p = A0 + (size_t)row * lda + k0 + koff;
  BF16x16 r;
#pragma unroll
  for (int e = 0; e < 8; ++e) r.u[e] = f2bf(p[e]);
#pragma unroll
  for (int e = 0; e < 8; ++e) r.u[8 + e] = f2bf(p[16 + e]);
  return r.v;
}

// B fragment from transposed bf16 weights Bt[N][K]; lane L: col = L&15, same K map.
__device__ inline v16bf load_b_t(const unsigned short* __restrict__ B0, int ldk,
                                 int k0, int lane) {
  int col = lane & 15, koff = (lane >> 4) * 8;
  const unsigned short* p = B0 + (size_t)col * ldk + k0 + koff;
  BF16x16 r;
  r.q[0] = *(const uint4*)(p);
  r.q[1] = *(const uint4*)(p + 16);
  return r.v;
}

// A fragment from bf16 LDS tile (row stride ldk ushorts).
__device__ inline v16bf load_a_lds(const unsigned short* h0, int ldk, int k0,
                                   int lane) {
  int row = lane & 15, koff = (lane >> 4) * 8;
  const unsigned short* p = h0 + row * ldk + k0 + koff;
  BF16x16 r;
#pragma unroll
  for (int e = 0; e < 8; ++e) { r.u[e] = p[e]; r.u[8 + e] = p[16 + e]; }
  return r.v;
}

__device__ inline float geluf(float x) {
  float x3 = x * x * x;
  return 0.5f * x * (1.0f + fast_tanh(0.7978845608028654f * (x + 0.044715f * x3)));
}
__device__ inline float sigmf(float x) { return fast_rcp(1.0f + __expf(-x)); }
__device__ inline float siluf(float x) { return x * sigmf(x); }

// ---------------------------------------------------------------------------
// Weight convert + transpose: W[K,N] f32 -> Bt[Npad,K] bf16 (zero padded cols)
// ---------------------------------------------------------------------------
__global__ void cvt_wt_kernel(const float* __restrict__ W,
                              unsigned short* __restrict__ Bt, int K, int N,
                              int Npad) {
  int idx = blockIdx.x * 256 + threadIdx.x;
  if (idx >= Npad * K) return;
  int n = idx / K, k = idx - n * K;
  float v = (n < N) ? W[(size_t)k * N + n] : 0.0f;
  Bt[idx] = f2bf(v);
}

// ---------------------------------------------------------------------------
// Generic WMMA GEMM: C = act(A[M,K] @ B[K,N] + bias) (+res).
// 4 waves / block; wave w handles N-tile (blockIdx.y*4 + w).
// ---------------------------------------------------------------------------
__global__ __launch_bounds__(128) void gemm_wmma_kernel(
    const float* __restrict__ A, int lda, const unsigned short* __restrict__ Bt,
    const float* __restrict__ bias, const float* __restrict__ res,
    float* __restrict__ C, int ldc, int K, int act) {
  int wv = threadIdx.x >> 5, lane = threadIdx.x & 31;
  int mt = blockIdx.x, nt = blockIdx.y * 4 + wv;
  const float* A0 = A + (size_t)mt * 16 * lda;
  const unsigned short* B0 = Bt + (size_t)nt * 16 * K;
  v8f acc = {};
  for (int k0 = 0; k0 < K; k0 += 32) {
    v16bf a = load_a_f32(A0, lda, k0, lane);
    v16bf b = load_b_t(B0, K, k0, lane);
    acc = wmma_bf16(a, b, acc);
  }
  int n = nt * 16 + (lane & 15);
  int m0 = mt * 16 + (lane >> 4) * 8;
  float bv = bias ? bias[n] : 0.0f;
#pragma unroll
  for (int g = 0; g < 8; ++g) {
    float v = acc[g] + bv;
    if (act == 1) v = geluf(v);
    else if (act == 2) v = siluf(v);
    else if (act == 3) v = sigmf(v);
    size_t o = (size_t)(m0 + g) * ldc + n;
    if (res) v += res[o];
    C[o] = v;
  }
}

// ---------------------------------------------------------------------------
// LayerNorm over D=128 (optionally of x1+x2). One block of 128 per row.
// ---------------------------------------------------------------------------
__global__ void ln_kernel(const float* __restrict__ x1,
                          const float* __restrict__ x2,
                          const float* __restrict__ g,
                          const float* __restrict__ bta,
                          float* __restrict__ out) {
  int row = blockIdx.x, t = threadIdx.x;
  size_t idx = (size_t)row * 128 + t;
  float v = x1[idx];
  if (x2) v += x2[idx];
  __shared__ float s1[128], s2[128];
  s1[t] = v; s2[t] = v * v; __syncthreads();
  for (int s = 64; s > 0; s >>= 1) {
    if (t < s) { s1[t] += s1[t + s]; s2[t] += s2[t + s]; }
    __syncthreads();
  }
  float mean = s1[0] * (1.0f / 128.0f);
  float var = s2[0] * (1.0f / 128.0f) - mean * mean;
  float inv = rsqrtf(var + 1e-5f);
  out[idx] = (v - mean) * inv * g[t] + bta[t];
}

// ---------------------------------------------------------------------------
// Fused pair kernel: per (b,i), stream j-tiles of 16 through 8 MLPs on WMMA,
// accumulate msg_s(+msg_s_from_v), delta_v, delta_t, three-body moments in LDS.
// p: 0=r0 1=r1 2=r2 3=r110 4..6=tb rad l  7=edge_proj (gelu, 8 outs)
// Staging of s_src/node_v tiles uses gfx1250 async global->LDS copies.
// ---------------------------------------------------------------------------
struct PairArgs {
  const float* rbf; const float* rhat; const float* mask;
  const float* s_src; const float* node_v;
  const unsigned short* w1t[8]; const float* b1[8];
  const unsigned short* w2t[8]; const float* b2[8];
  const float* coef;
  float* msg; float* dv; float* dt; float* feat; float* edge_bias;
};

__global__ __launch_bounds__(256) void pair_fused_kernel(PairArgs P) {
  __shared__ float sst[16][132];          // s_src tile  [j][d] (async-filled)
  __shared__ float vst[16][3][132];       // node_v tile [j][m][d] (async-filled)
  __shared__ float Y1s[16][3];
  __shared__ float Y2s[16][5];
  __shared__ float mks[16];
  __shared__ unsigned short hidb[16][136];  // hidden exchange (bf16)
  __shared__ float acc_s[128];
  __shared__ float acc_v[3][128];
  __shared__ float acc_t[5][128];
  __shared__ float acc_a[9][128];

  const int tid = threadIdx.x;
  const int bi = blockIdx.x, b = bi >> 7, i = bi & 127;
  if (tid < 128) {
    acc_s[tid] = 0.f;
    for (int m = 0; m < 3; ++m) acc_v[m][tid] = 0.f;
    for (int m = 0; m < 5; ++m) acc_t[m][tid] = 0.f;
    for (int m = 0; m < 9; ++m) acc_a[m][tid] = 0.f;
  }
  __syncthreads();

  const int w = tid >> 5, lane = tid & 31;
  const int n = lane & 15, mb8 = (lane >> 4) * 8;

  for (int jt = 0; jt < 8; ++jt) {
    int j0 = jt * 16;
    // ---- async stage: s_src tile (512 x 16B) + node_v tile (1536 x 16B) ----
#pragma unroll
    for (int c = tid; c < 512; c += 256) {
      int jj = c >> 5, d4 = (c & 31) * 4;
      async_g2l_b128(&sst[jj][d4],
                     P.s_src + ((size_t)(b * 128 + j0 + jj)) * 128 + d4);
    }
#pragma unroll
    for (int c = tid; c < 1536; c += 256) {
      int jj = c / 96, rem = c - jj * 96, m = rem >> 5, d4 = (rem & 31) * 4;
      async_g2l_b128(&vst[jj][m][d4],
                     P.node_v + (((size_t)(b * 128 + j0 + jj)) * 3 + m) * 128 + d4);
    }
    if (tid < 16) {
      int jj = tid;
      size_t e = ((size_t)(b * 128 + i)) * 128 + (j0 + jj);
      float x = P.rhat[e * 3 + 0], y = P.rhat[e * 3 + 1], z = P.rhat[e * 3 + 2];
      const float s3 = 1.7320508075688772f;     // sqrt(3)
      const float c15 = 3.872983346207417f;     // sqrt(15)
      const float c5h = 1.118033988749895f;     // sqrt(5)/2
      Y1s[jj][0] = s3 * x; Y1s[jj][1] = s3 * y; Y1s[jj][2] = s3 * z;
      Y2s[jj][0] = c15 * x * y;
      Y2s[jj][1] = c15 * y * z;
      Y2s[jj][2] = c5h * (2.f * z * z - x * x - y * y);
      Y2s[jj][3] = c15 * x * z;
      Y2s[jj][4] = 0.5f * c15 * (x * x - y * y);
      mks[jj] = P.mask[e];
    }
    // prefetch next rbf tile into cache while we wait
    if (jt + 1 < 8) {
      const float* pn = P.rbf +
          (((size_t)(b * 128 + i)) * 128 + j0 + 16 + (lane & 15)) * 32;
      __builtin_prefetch(pn, 0, 1);
    }
    wait_async0();
    __syncthreads();

    const float* a0 = P.rbf + (((size_t)(b * 128 + i)) * 128 + j0) * 32;
    v16bf a_rbf = load_a_f32(a0, 32, 0, lane);

    for (int p = 0; p < 8; ++p) {
      // hidden = act(rbf @ W1 + b1); wave w owns hidden cols [16w,16w+16)
      v16bf bw = load_b_t(P.w1t[p] + (size_t)w * 16 * 32, 32, 0, lane);
      v8f zc = {};
      v8f hc = wmma_bf16(a_rbf, bw, zc);
      int hcol = w * 16 + n;
      float b1v = P.b1[p][hcol];
#pragma unroll
      for (int g = 0; g < 8; ++g) {
        float hv = hc[g] + b1v;
        hv = (p == 7) ? geluf(hv) : siluf(hv);
        hidb[mb8 + g][hcol] = f2bf(hv);
      }
      __syncthreads();
      // out = hidden @ W2 + b2 (K=128); wave w owns out cols [16w,16w+16)
      v8f oc = {};
      float b2v = 0.f;
      bool active = (p < 7) || (w == 0);
      if (active) {
#pragma unroll
        for (int kc = 0; kc < 4; ++kc) {
          v16bf ah = load_a_lds(&hidb[0][0], 136, kc * 32, lane);
          v16bf bw2 = load_b_t(P.w2t[p] + (size_t)w * 16 * 128, 128, kc * 32, lane);
          oc = wmma_bf16(ah, bw2, oc);
        }
        if (p < 7) b2v = P.b2[p][w * 16 + n];
        else b2v = (n < 8) ? P.b2[p][n] : 0.f;
      }
      int d = w * 16 + n;
      if (p == 0) {
        float s = 0;
#pragma unroll
        for (int g = 0; g < 8; ++g) { int jj = mb8 + g; s += (oc[g] + b2v) * sst[jj][d] * mks[jj]; }
        atomicAdd(&acc_s[d], s);
      } else if (p == 1) {
        float s0 = 0, s1 = 0, s2 = 0;
#pragma unroll
        for (int g = 0; g < 8; ++g) {
          int jj = mb8 + g;
          float hh = (oc[g] + b2v) * sst[jj][d] * mks[jj];
          s0 += hh * Y1s[jj][0]; s1 += hh * Y1s[jj][1]; s2 += hh * Y1s[jj][2];
        }
        atomicAdd(&acc_v[0][d], s0); atomicAdd(&acc_v[1][d], s1); atomicAdd(&acc_v[2][d], s2);
      } else if (p == 2) {
        float s[5] = {0, 0, 0, 0, 0};
#pragma unroll
        for (int g = 0; g < 8; ++g) {
          int jj = mb8 + g;
          float hh = (oc[g] + b2v) * sst[jj][d] * mks[jj];
#pragma unroll
          for (int m = 0; m < 5; ++m) s[m] += hh * Y2s[jj][m];
        }
        for (int m = 0; m < 5; ++m) atomicAdd(&acc_t[m][d], s[m]);
      } else if (p == 3) {
        float s = 0;
#pragma unroll
        for (int g = 0; g < 8; ++g) {
          int jj = mb8 + g;
          float dot = vst[jj][0][d] * Y1s[jj][0] + vst[jj][1][d] * Y1s[jj][1] +
                      vst[jj][2][d] * Y1s[jj][2];
          s += (oc[g] + b2v) * dot * mks[jj];
        }
        atomicAdd(&acc_s[d], s);
      } else if (p == 4) {
        float s = 0;
#pragma unroll
        for (int g = 0; g < 8; ++g) { int jj = mb8 + g; s += (oc[g] + b2v) * mks[jj]; }
        atomicAdd(&acc_a[0][d], s);
      } else if (p == 5) {
        float s0 = 0, s1 = 0, s2 = 0;
#pragma unroll
        for (int g = 0; g < 8; ++g) {
          int jj = mb8 + g;
          float ww = (oc[g] + b2v) * mks[jj];
          s0 += ww * Y1s[jj][0]; s1 += ww * Y1s[jj][1]; s2 += ww * Y1s[jj][2];
        }
        atomicAdd(&acc_a[1][d], s0); atomicAdd(&acc_a[2][d], s1); atomicAdd(&acc_a[3][d], s2);
      } else if (p == 6) {
        float s[5] = {0, 0, 0, 0, 0};
#pragma unroll
        for (int g = 0; g < 8; ++g) {
          int jj = mb8 + g;
          float ww = (oc[g] + b2v) * mks[jj];
#pragma unroll
          for (int m = 0; m < 5; ++m) s[m] += ww * Y2s[jj][m];
        }
        for (int m = 0; m < 5; ++m) atomicAdd(&acc_a[4 + m][d], s[m]);
      } else {  // p == 7: edge bias -> [b,i,j,h]
        if (w == 0 && n < 8) {
#pragma unroll
          for (int g = 0; g < 8; ++g) {
            int jj = mb8 + g;
            P.edge_bias[(((size_t)(b * 128 + i)) * 128 + (j0 + jj)) * 8 + n] =
                oc[g] + b2v;
          }
        }
      }
      __syncthreads();
    }
  }

  if (tid < 128) {
    int d = tid;
    size_t base = (size_t)(b * 128 + i) * 128;
    float c0 = P.coef[0], c1 = P.coef[1], c2 = P.coef[2];
    float f = c0 * acc_a[0][d] * acc_a[0][d];
    float t1 = 0, t2 = 0;
    for (int m = 1; m < 4; ++m) t1 += acc_a[m][d] * acc_a[m][d];
    for (int m = 4; m < 9; ++m) t2 += acc_a[m][d] * acc_a[m][d];
    P.feat[base + d] = f + c1 * t1 + c2 * t2;
    P.msg[base + d] = acc_s[d];
    for (int m = 0; m < 3; ++m) P.dv[((size_t)(b * 128 + i) * 3 + m) * 128 + d] = acc_v[m][d];
    for (int m = 0; m < 5; ++m) P.dt[((size_t)(b * 128 + i) * 5 + m) * 128 + d] = acc_t[m][d];
  }
}

// ---------------------------------------------------------------------------
// Attention small kernels
// ---------------------------------------------------------------------------
__global__ void logits_kernel(const float* __restrict__ q,
                              const float* __restrict__ k,
                              const float* __restrict__ eb,
                              float* __restrict__ logits,
                              float* __restrict__ out_eb) {
  int bhi = blockIdx.x, j = threadIdx.x;
  int i = bhi & 127, bh = bhi >> 7, h = bh & 7, b = bh >> 3;
  const float* qp = q + ((size_t)(b * 128 + i)) * 128 + h * 16;
  const float* kp = k + ((size_t)(b * 128 + j)) * 128 + h * 16;
  float s = 0;
#pragma unroll
  for (int c = 0; c < 16; ++c) s += qp[c] * kp[c];
  s = s * 0.25f + eb[(((size_t)(b * 128 + i)) * 128 + j) * 8 + h];
  logits[(size_t)bhi * 128 + j] = s;
  out_eb[(((size_t)(b * 128 + i)) * 128 + j) * 8 + h] = s;
}

__global__ void softmax_kernel(float* __restrict__ logits) {
  int row = blockIdx.x, t = threadIdx.x;
  float v = logits[(size_t)row * 128 + t];
  __shared__ float sb[128];
  sb[t] = v; __syncthreads();
  for (int s = 64; s > 0; s >>= 1) { if (t < s) sb[t] = fmaxf(sb[t], sb[t + s]); __syncthreads(); }
  float mx = sb[0]; __syncthreads();
  float e = __expf(v - mx);
  sb[t] = e; __syncthreads();
  for (int s = 64; s > 0; s >>= 1) { if (t < s) sb[t] += sb[t + s]; __syncthreads(); }
  logits[(size_t)row * 128 + t] = e * fast_rcp(sb[0]);
}

__global__ void attnv_kernel(const float* __restrict__ probs,
                             const float* __restrict__ v, float* __restrict__ o) {
  int bi = blockIdx.x, t = threadIdx.x;
  int b = bi >> 7, i = bi & 127, h = t >> 4;
  const float* pr = probs + (((size_t)(b * 8 + h)) * 128 + i) * 128;
  float s = 0;
  for (int j = 0; j < 128; ++j) s += pr[j] * v[((size_t)(b * 128 + j)) * 128 + t];
  o[(size_t)bi * 128 + t] = s;
}

// ---------------------------------------------------------------------------
// Norms, concat builders, fused outputs
// ---------------------------------------------------------------------------
__global__ void norms_kernel(const float* __restrict__ nv,
                             const float* __restrict__ nt,
                             float* __restrict__ vn, float* __restrict__ tn) {
  int idx = blockIdx.x * 256 + threadIdx.x;
  if (idx >= 512 * 128) return;
  int p = idx >> 7, d = idx & 127;
  float s = 0;
  for (int m = 0; m < 3; ++m) { float x = nv[((size_t)p * 3 + m) * 128 + d]; s += x * x; }
  vn[idx] = sqrtf(s);
  s = 0;
  for (int m = 0; m < 5; ++m) { float x = nt[((size_t)p * 5 + m) * 128 + d]; s += x * x; }
  tn[idx] = sqrtf(s);
}

__global__ void gatein_kernel(const float* dsa, const float* dsb, const float* dsc,
                              const float* vn, const float* tn, float* gi) {
  int idx = blockIdx.x * 256 + threadIdx.x;
  if (idx >= 512 * 128) return;
  int p = idx >> 7, d = idx & 127;
  float* r = gi + (size_t)p * 384;
  r[d] = dsa[idx];
  r[128 + d] = dsb[idx] + dsc[idx];
  r[256 + d] = vn[idx] + tn[idx];
}

__global__ void cat2_kernel(const float* a, const float* b, float* out) {
  int idx = blockIdx.x * 256 + threadIdx.x;
  if (idx >= 512 * 128) return;
  int p = idx >> 7, d = idx & 127;
  out[(size_t)p * 256 + d] = a[idx];
  out[(size_t)p * 256 + 128 + d] = b[idx];
}

__global__ void fuseh_kernel(const float* node_s, const float* gs,
                             const float* dsa, const float* dsb, const float* dsc,
                             const float* lng, const float* lnb, float* hnew) {
  int row = blockIdx.x, t = threadIdx.x;
  size_t idx = (size_t)row * 128 + t;
  float v = gs[idx] * (dsa[idx] + dsb[idx] + dsc[idx]);
  __shared__ float s1[128], s2[128];
  s1[t] = v; s2[t] = v * v; __syncthreads();
  for (int s = 64; s > 0; s >>= 1) {
    if (t < s) { s1[t] += s1[t + s]; s2[t] += s2[t + s]; }
    __syncthreads();
  }
  float mean = s1[0] * (1.0f / 128.0f);
  float var = s2[0] * (1.0f / 128.0f) - mean * mean;
  float inv = rsqrtf(var + 1e-5f);
  hnew[idx] = node_s[idx] + ((v - mean) * inv * lng[t] + lnb[t]);
}

__global__ void fusevt_kernel(const float* nvt, const float* gate, const float* dlt,
                              const float* sc1, const float* sc2,
                              float* out, int M) {
  int idx = blockIdx.x * 256 + threadIdx.x;
  if (idx >= 512 * M * 128) return;
  int d = idx & 127, pm = idx >> 7, p = pm / M;
  out[idx] = nvt[idx] + gate[(size_t)p * 128 + d] * dlt[idx] * sc1[d] * sc2[d];
}

// ---------------------------------------------------------------------------
// Input index map (top-level dict insertion order; params flattened as JAX
// pytree: dict keys sorted, lists in order).
// ---------------------------------------------------------------------------
enum {
  IN_NODE_S = 0, IN_NODE_V, IN_NODE_T, IN_RBF, IN_RHAT, IN_MASK,
  A_BFF1 = 6, A_BFF2, A_BK, A_BO, A_BQ, A_BV, A_LN1B, A_LN1G, A_LN2B, A_LN2G,
  A_WFF1, A_WFF2, A_WK, A_WO, A_WQ, A_WV,                  // 6..21
  E_B1 = 22, E_B2, E_W1, E_W2,                             // 22..25
  F_GSB = 26, F_GSW, F_GTB, F_GTW, F_GVB, F_GVW, F_LNB, F_LNG, F_SCT, F_SCV,
  S_OUTSB = 36, S_OUTSLNB, S_OUTSLNG, S_OUTSW,
  S_R0B1 = 40, S_R0B2, S_R0W1, S_R0W2,
  S_R1B1 = 44, S_R1B2, S_R1W1, S_R1W2,
  S_R110B1 = 48, S_R110B2, S_R110W1, S_R110W2,
  S_R2B1 = 52, S_R2B2, S_R2W1, S_R2W2,
  S_SRCW = 56, S_TSC, S_VSC,
  T_COEF = 59, T_OUTB, T_OUTW,
  T_R0B1 = 62, T_R0B2, T_R0W1, T_R0W2,
  T_R1B1 = 66, T_R1B2, T_R1W1, T_R1W2,
  T_R2B1 = 70, T_R2B2, T_R2W1, T_R2W2
};

extern "C" void kernel_launch(void* const* d_in, const int* in_sizes, int n_in,
                              void* d_out, int out_size, void* d_ws, size_t ws_size,
                              hipStream_t stream) {
  (void)in_sizes; (void)n_in; (void)out_size; (void)ws_size;
  const int P = 512;  // B*N

  auto F = [&](int i) { return (const float*)d_in[i]; };

  // ---- workspace bump allocator (256B aligned) ----
  char* wp = (char*)d_ws;
  auto alloc = [&](size_t bytes) -> void* {
    void* p = (void*)wp;
    wp += (bytes + 255) & ~(size_t)255;
    return p;
  };
  float* s_src   = (float*)alloc(P * 128 * 4);
  float* h_ln    = (float*)alloc(P * 128 * 4);
  float* qb      = (float*)alloc(P * 128 * 4);
  float* kb      = (float*)alloc(P * 128 * 4);
  float* vb      = (float*)alloc(P * 128 * 4);
  float* logits  = (float*)alloc((size_t)4 * 8 * 128 * 128 * 4);
  float* attn_c  = (float*)alloc(P * 128 * 4);
  float* ob      = (float*)alloc(P * 128 * 4);
  float* h2      = (float*)alloc(P * 128 * 4);
  float* ffn_h   = (float*)alloc((size_t)P * 512 * 4);
  float* dsa     = (float*)alloc(P * 128 * 4);
  float* msg     = (float*)alloc(P * 128 * 4);
  float* msgln   = (float*)alloc(P * 128 * 4);
  float* dvb     = (float*)alloc(P * 3 * 128 * 4);
  float* dtb     = (float*)alloc(P * 5 * 128 * 4);
  float* featb   = (float*)alloc(P * 128 * 4);
  float* dsb     = (float*)alloc(P * 128 * 4);
  float* dsc     = (float*)alloc(P * 128 * 4);
  float* edge_b  = (float*)alloc((size_t)4 * 128 * 128 * 8 * 4);
  float* vn      = (float*)alloc(P * 128 * 4);
  float* tn      = (float*)alloc(P * 128 * 4);
  float* gate_in = (float*)alloc((size_t)P * 384 * 4);
  float* gvin    = (float*)alloc((size_t)P * 256 * 4);
  float* gtin    = (float*)alloc((size_t)P * 256 * 4);
  float* gsv     = (float*)alloc(P * 128 * 4);
  float* gvv     = (float*)alloc(P * 128 * 4);
  float* gtv     = (float*)alloc(P * 128 * 4);

  auto allocw = [&](size_t elems) { return (unsigned short*)alloc(elems * 2); };
  unsigned short* wq_t   = allocw(128 * 128);
  unsigned short* wk_t   = allocw(128 * 128);
  unsigned short* wv_t   = allocw(128 * 128);
  unsigned short* wo_t   = allocw(128 * 128);
  unsigned short* wff1_t = allocw((size_t)512 * 128);
  unsigned short* wff2_t = allocw((size_t)128 * 512);
  unsigned short* src_t  = allocw(128 * 128);
  unsigned short* outs_t = allocw(128 * 128);
  unsigned short* outw_t = allocw(128 * 128);
  unsigned short* gs_t   = allocw((size_t)128 * 384);
  unsigned short* gv_t   = allocw((size_t)128 * 256);
  unsigned short* gt_t   = allocw((size_t)128 * 256);
  unsigned short* w1t[8]; unsigned short* w2t[8];
  for (int p = 0; p < 8; ++p) {
    w1t[p] = allocw(128 * 32);
    w2t[p] = allocw((p == 7 ? 16 : 128) * 128);
  }

  auto cvt = [&](int wi, unsigned short* Bt, int K, int N, int Npad) {
    int tot = Npad * K;
    cvt_wt_kernel<<<(tot + 255) / 256, 256, 0, stream>>>(F(wi), Bt, K, N, Npad);
  };
  cvt(A_WQ, wq_t, 128, 128, 128);
  cvt(A_WK, wk_t, 128, 128, 128);
  cvt(A_WV, wv_t, 128, 128, 128);
  cvt(A_WO, wo_t, 128, 128, 128);
  cvt(A_WFF1, wff1_t, 128, 512, 512);
  cvt(A_WFF2, wff2_t, 512, 128, 128);
  cvt(S_SRCW, src_t, 128, 128, 128);
  cvt(S_OUTSW, outs_t, 128, 128, 128);
  cvt(T_OUTW, outw_t, 128, 128, 128);
  cvt(F_GSW, gs_t, 384, 128, 128);
  cvt(F_GVW, gv_t, 256, 128, 128);
  cvt(F_GTW, gt_t, 256, 128, 128);
  // MLP order: r0, r1, r2, r110, rad0, rad1, rad2, edge_proj
  const int mw1[8] = {S_R0W1, S_R1W1, S_R2W1, S_R110W1, T_R0W1, T_R1W1, T_R2W1, E_W1};
  const int mw2[8] = {S_R0W2, S_R1W2, S_R2W2, S_R110W2, T_R0W2, T_R1W2, T_R2W2, E_W2};
  const int mb1[8] = {S_R0B1, S_R1B1, S_R2B1, S_R110B1, T_R0B1, T_R1B1, T_R2B1, E_B1};
  const int mb2[8] = {S_R0B2, S_R1B2, S_R2B2, S_R110B2, T_R0B2, T_R1B2, T_R2B2, E_B2};
  for (int p = 0; p < 8; ++p) {
    cvt(mw1[p], w1t[p], 32, 128, 128);
    cvt(mw2[p], w2t[p], 128, (p == 7 ? 8 : 128), (p == 7 ? 16 : 128));
  }

  auto gemm = [&](const float* A, int lda, const unsigned short* Bt,
                  const float* bias, const float* res, float* C, int ldc,
                  int M, int Ncols, int K, int act) {
    gemm_wmma_kernel<<<dim3(M / 16, Ncols / 64), dim3(128), 0, stream>>>(
        A, lda, Bt, bias, res, C, ldc, K, act);
  };

  float* out_h = (float*)d_out;                 // [B,N,D]
  float* out_v = out_h + 65536;                 // [B,N,3,D]
  float* out_t = out_v + 196608;                // [B,N,5,D]
  float* out_e = out_t + 327680;                // [B,N,N,H]

  // ---- node prep ----
  norms_kernel<<<(P * 128 + 255) / 256, 256, 0, stream>>>(F(IN_NODE_V), F(IN_NODE_T), vn, tn);
  ln_kernel<<<P, 128, 0, stream>>>(F(IN_NODE_S), nullptr, F(A_LN1G), F(A_LN1B), h_ln);
  gemm(F(IN_NODE_S), 128, src_t, nullptr, nullptr, s_src, 128, P, 128, 128, 0);

  // ---- fused pair path (radials + three-body + edge_proj), all WMMA ----
  PairArgs pa;
  pa.rbf = F(IN_RBF); pa.rhat = F(IN_RHAT); pa.mask = F(IN_MASK);
  pa.s_src = s_src; pa.node_v = F(IN_NODE_V);
  for (int p = 0; p < 8; ++p) {
    pa.w1t[p] = w1t[p]; pa.w2t[p] = w2t[p];
    pa.b1[p] = F(mb1[p]); pa.b2[p] = F(mb2[p]);
  }
  pa.coef = F(T_COEF);
  pa.msg = msg; pa.dv = dvb; pa.dt = dtb; pa.feat = featb; pa.edge_bias = edge_b;
  pair_fused_kernel<<<P, 256, 0, stream>>>(pa);

  // ---- attention path ----
  gemm(h_ln, 128, wq_t, F(A_BQ), nullptr, qb, 128, P, 128, 128, 0);
  gemm(h_ln, 128, wk_t, F(A_BK), nullptr, kb, 128, P, 128, 128, 0);
  gemm(h_ln, 128, wv_t, F(A_BV), nullptr, vb, 128, P, 128, 128, 0);
  logits_kernel<<<4 * 8 * 128, 128, 0, stream>>>(qb, kb, edge_b, logits, out_e);
  softmax_kernel<<<4 * 8 * 128, 128, 0, stream>>>(logits);
  attnv_kernel<<<P, 128, 0, stream>>>(logits, vb, attn_c);
  gemm(attn_c, 128, wo_t, F(A_BO), nullptr, ob, 128, P, 128, 128, 0);
  ln_kernel<<<P, 128, 0, stream>>>(F(IN_NODE_S), ob, F(A_LN2G), F(A_LN2B), h2);
  gemm(h2, 128, wff1_t, F(A_BFF1), nullptr, ffn_h, 512, P, 512, 128, 1 /*gelu*/);
  gemm(ffn_h, 512, wff2_t, F(A_BFF2), ob, dsa, 128, P, 128, 512, 0);  // delta_sa

  // ---- SE(3) scalar output & three-body output ----
  ln_kernel<<<P, 128, 0, stream>>>(msg, nullptr, F(S_OUTSLNG), F(S_OUTSLNB), msgln);
  gemm(msgln, 128, outs_t, F(S_OUTSB), nullptr, dsb, 128, P, 128, 128, 0);
  gemm(featb, 128, outw_t, F(T_OUTB), nullptr, dsc, 128, P, 128, 128, 0);

  // ---- fusion ----
  gatein_kernel<<<(P * 128 + 255) / 256, 256, 0, stream>>>(dsa, dsb, dsc, vn, tn, gate_in);
  gemm(gate_in, 384, gs_t, F(F_GSB), nullptr, gsv, 128, P, 128, 384, 3 /*sigmoid*/);
  fuseh_kernel<<<P, 128, 0, stream>>>(F(IN_NODE_S), gsv, dsa, dsb, dsc,
                                      F(F_LNG), F(F_LNB), out_h);
  cat2_kernel<<<(P * 128 + 255) / 256, 256, 0, stream>>>(out_h, vn, gvin);
  gemm(gvin, 256, gv_t, F(F_GVB), nullptr, gvv, 128, P, 128, 256, 3);
  fusevt_kernel<<<(P * 3 * 128 + 255) / 256, 256, 0, stream>>>(
      F(IN_NODE_V), gvv, dvb, F(S_VSC), F(F_SCV), out_v, 3);
  cat2_kernel<<<(P * 128 + 255) / 256, 256, 0, stream>>>(out_h, tn, gtin);
  gemm(gtin, 256, gt_t, F(F_GTB), nullptr, gtv, 128, P, 128, 256, 3);
  fusevt_kernel<<<(P * 5 * 128 + 255) / 256, 256, 0, stream>>>(
      F(IN_NODE_T), gtv, dtb, F(S_TSC), F(F_SCT), out_t, 5);
}
